// DEQ_MLP_35966056137092
// MI455X (gfx1250) — compile-verified, matrix operand order checked by reading, not run on hardware
//
#include <hip/hip_runtime.h>
#include <hip/hip_bf16.h>

typedef __attribute__((ext_vector_type(8)))  float  v8f;
typedef __attribute__((ext_vector_type(16))) __bf16 v16bf;
typedef int v4i __attribute__((vector_size(16)));
typedef __attribute__((address_space(1))) v4i GV4;   // global int4
typedef __attribute__((address_space(3))) v4i LV4;   // LDS int4

union FragU { uint4 q[2]; v16bf v; };

#define BM   64
#define BN   128
#define BK   64
#define LDSP 72   // padded K-stride (elements): 144B rows, 16B aligned, bank-skewed

// ---- CDNA5 async global->LDS staging (guarded; sync fallback) --------------
#if defined(__gfx1250__) && \
    __has_builtin(__builtin_amdgcn_global_load_async_to_lds_b128) && \
    __has_builtin(__builtin_amdgcn_s_wait_asynccnt)
#define DEQ_ASYNC_LDS 1
#else
#define DEQ_ASYNC_LDS 0
#endif

__device__ __forceinline__ void copy16_g2l(const __hip_bfloat16* __restrict__ gsrc,
                                           unsigned short* __restrict__ ldst) {
#if DEQ_ASYNC_LDS
  __builtin_amdgcn_global_load_async_to_lds_b128((GV4*)gsrc, (LV4*)ldst, 0, 0);
#else
  *(uint4*)ldst = *(const uint4*)gsrc;
#endif
}

__device__ __forceinline__ void staging_fence() {
#if DEQ_ASYNC_LDS
  __builtin_amdgcn_s_wait_asynccnt(0);
#endif
  __syncthreads();
}

// ---------------------------------------------------------------------------
// bf16 WMMA GEMM:  C[M,N] = A[M,K] @ Bt[N,K]^T + bias.
// A row-major MxK, Bt row-major NxK (weights pre-transposed once per launch).
// 64x128 block tile, 8 waves (2x4), wave = 32x32 patch = 2x2 WMMA tiles,
// BK=64 -> 8 v_wmma_f32_16x16x32_bf16 per wave per LDS stage.
// ---------------------------------------------------------------------------
template <int RELU, int WF, int WB>
__global__ __launch_bounds__(256) void deq_gemm_bf16_wmma(
    int M, int N, int K,
    const __hip_bfloat16* __restrict__ A,
    const __hip_bfloat16* __restrict__ Bt,
    const float*          __restrict__ bias,
    float*                __restrict__ Cf,
    __hip_bfloat16*       __restrict__ Cb)
{
  __shared__ __align__(16) unsigned short As[BM * LDSP];
  __shared__ __align__(16) unsigned short Bs[BN * LDSP];

  const int tid  = threadIdx.x;
  const int lane = tid & 31;
  const int wave = tid >> 5;
  const int wr   = wave & 1;    // 2 wave-rows  (32 rows each)
  const int wc   = wave >> 1;   // 4 wave-cols  (32 cols each)
  const int m0   = blockIdx.y * BM;
  const int n0   = blockIdx.x * BN;

  v8f acc[2][2];
  #pragma unroll
  for (int i = 0; i < 2; ++i)
    #pragma unroll
    for (int j = 0; j < 2; ++j)
      #pragma unroll
      for (int e = 0; e < 8; ++e) acc[i][j][e] = 0.f;

  const int kbase = (lane >> 4) * 8;  // 16-bit A/B layout: lane<16 -> K 0..7/16..23
  const int l16   = lane & 15;

  for (int k0 = 0; k0 < K; k0 += BK) {
    // ---- stage A (BM x BK) and B (BN x BK), 16B chunks, async when available
    #pragma unroll
    for (int i = 0; i < 2; ++i) {               // 64*64/8 = 512 chunks
      const int c = tid + i * 256;
      const int r = c >> 3, c8 = c & 7;
      copy16_g2l(A + (size_t)(m0 + r) * K + k0 + c8 * 8, &As[r * LDSP + c8 * 8]);
    }
    #pragma unroll
    for (int i = 0; i < 4; ++i) {               // 128*64/8 = 1024 chunks
      const int c = tid + i * 256;
      const int r = c >> 3, c8 = c & 7;
      copy16_g2l(Bt + (size_t)(n0 + r) * K + k0 + c8 * 8, &Bs[r * LDSP + c8 * 8]);
    }
    // prefetch next K tile into cache while WMMAs run
    if (k0 + BK < K) {
      __builtin_prefetch(A  + (size_t)(m0 + (tid & 63))  * K + k0 + BK, 0, 1);
      __builtin_prefetch(Bt + (size_t)(n0 + (tid & 127)) * K + k0 + BK, 0, 1);
    }
    staging_fence();

    // ---- two 32-wide K windows per stage ----
    #pragma unroll
    for (int w = 0; w < 2; ++w) {
      const int kb = w * 32 + kbase;
      FragU a[2], b[2];
      #pragma unroll
      for (int mt = 0; mt < 2; ++mt) {
        const int r = wr * 32 + mt * 16 + l16;
        a[mt].q[0] = *(const uint4*)(&As[r * LDSP + kb]);
        a[mt].q[1] = *(const uint4*)(&As[r * LDSP + kb + 16]);
      }
      #pragma unroll
      for (int nt = 0; nt < 2; ++nt) {
        const int n = wc * 32 + nt * 16 + l16;
        b[nt].q[0] = *(const uint4*)(&Bs[n * LDSP + kb]);
        b[nt].q[1] = *(const uint4*)(&Bs[n * LDSP + kb + 16]);
      }
      #pragma unroll
      for (int mt = 0; mt < 2; ++mt)
        #pragma unroll
        for (int nt = 0; nt < 2; ++nt)
          acc[mt][nt] = __builtin_amdgcn_wmma_f32_16x16x32_bf16(
              false, a[mt].v, false, b[nt].v, (short)0, acc[mt][nt], false, false);
    }
    __syncthreads();
  }

  // ---- epilogue (branch-free; C/D layout: VGPR i -> M=(lane/16)*8+i, N=lane%16)
  const int rowHalf = (lane >> 4) * 8;
  #pragma unroll
  for (int mt = 0; mt < 2; ++mt) {
    #pragma unroll
    for (int nt = 0; nt < 2; ++nt) {
      const int gcol = n0 + wc * 32 + nt * 16 + l16;
      const float bv = bias[gcol];
      #pragma unroll
      for (int i = 0; i < 8; ++i) {
        const int grow = m0 + wr * 32 + mt * 16 + rowHalf + i;
        float v = acc[mt][nt][i] + bv;
        if (RELU) v = fmaxf(v, 0.f);
        const size_t off = (size_t)grow * N + gcol;
        if (WF) Cf[off] = v;
        if (WB) Cb[off] = __float2bfloat16(v);
      }
    }
  }
}

// ---------------------------------------------------------------------------
// Anderson step: one wave per batch row. Gram(6x6) over D, shfl_xor reduce,
// redundant per-lane bordered 7x7 Gauss-Jordan solve, xk = alpha . F written
// to X slot (f32) and zb (bf16, next GEMM A operand).
// ---------------------------------------------------------------------------
__device__ __forceinline__ int tri6(int i, int j) {
  if (i > j) { int t = i; i = j; j = t; }
  return i * 6 - (i * (i - 1)) / 2 + (j - i);
}

__global__ __launch_bounds__(256) void anderson_combine(
    const float* __restrict__ X, const float* __restrict__ F,
    float* __restrict__ Xslot, __hip_bfloat16* __restrict__ zb,
    int k, int B, int D)
{
  const int lane = threadIdx.x & 31;
  const int wave = threadIdx.x >> 5;
  const int row  = blockIdx.x * 8 + wave;
  if (row >= B) return;
  const int nv = (k < 6) ? k : 6;
  const size_t rd = (size_t)row * D;
  const size_t sl = (size_t)B * D;

  float gg[21];
  #pragma unroll
  for (int t = 0; t < 21; ++t) gg[t] = 0.f;

  for (int d = lane; d < D; d += 32) {
    float g[6];
    #pragma unroll
    for (int j = 0; j < 6; ++j)
      g[j] = (j < nv) ? (F[(size_t)j * sl + rd + d] - X[(size_t)j * sl + rd + d]) : 0.f;
    int t = 0;
    #pragma unroll
    for (int i = 0; i < 6; ++i)
      #pragma unroll
      for (int j = i; j < 6; ++j)
        gg[t++] += g[i] * g[j];
  }
  #pragma unroll
  for (int off = 16; off >= 1; off >>= 1)
    #pragma unroll
    for (int t = 0; t < 21; ++t)
      gg[t] += __shfl_xor(gg[t], off, 32);

  float Hm[7][8];
  for (int r = 0; r < 7; ++r)
    for (int c = 0; c < 8; ++c) Hm[r][c] = 0.f;
  for (int j = 0; j < 6; ++j) {
    const float vj = (j < nv) ? 1.f : 0.f;
    Hm[0][1 + j] = vj; Hm[1 + j][0] = vj;
  }
  for (int i = 0; i < 6; ++i)
    for (int j = 0; j < 6; ++j) {
      float v;
      if (i < nv && j < nv) v = gg[tri6(i, j)] + ((i == j) ? 1e-4f : 0.f);
      else                  v = (i == j) ? 1.f : 0.f;
      Hm[1 + i][1 + j] = v;
    }
  Hm[0][7] = 1.f;   // rhs = e0

  for (int c = 0; c < 7; ++c) {
    int piv = c; float best = fabsf(Hm[c][c]);
    for (int r = c + 1; r < 7; ++r) {
      const float a = fabsf(Hm[r][c]);
      if (a > best) { best = a; piv = r; }
    }
    if (piv != c)
      for (int cc = 0; cc < 8; ++cc) {
        const float t = Hm[c][cc]; Hm[c][cc] = Hm[piv][cc]; Hm[piv][cc] = t;
      }
    const float inv = 1.f / Hm[c][c];
    for (int r = 0; r < 7; ++r) {
      if (r == c) continue;
      const float f = Hm[r][c] * inv;
      for (int cc = c; cc < 8; ++cc) Hm[r][cc] -= f * Hm[c][cc];
    }
  }
  float alpha[6];
  for (int j = 0; j < 6; ++j)
    alpha[j] = (j < nv) ? Hm[1 + j][7] / Hm[1 + j][1 + j] : 0.f;

  for (int d = lane; d < D; d += 32) {
    float s = 0.f;
    for (int j = 0; j < 6; ++j)
      if (j < nv) s += alpha[j] * F[(size_t)j * sl + rd + d];
    Xslot[rd + d] = s;
    zb[rd + d]    = __float2bfloat16(s);
  }
}

// ---------------------------------------------------------------------------
// conversion kernels
// ---------------------------------------------------------------------------
__global__ void f32_to_bf16_kernel(const float* __restrict__ in,
                                   __hip_bfloat16* __restrict__ out, int n) {
  const int i = blockIdx.x * 256 + threadIdx.x;
  if (i < n) out[i] = __float2bfloat16(in[i]);
}

// in: K x N (f32, row-major)  ->  out: N x K (bf16, row-major) == in^T
__global__ void f32_to_bf16_transpose_kernel(const float* __restrict__ in,
                                             __hip_bfloat16* __restrict__ out,
                                             int K, int N) {
  const int i = blockIdx.x * 256 + threadIdx.x;
  if (i < K * N) {
    const int n = i / K, k = i % K;
    out[i] = __float2bfloat16(in[(size_t)k * N + n]);
  }
}

// ---------------------------------------------------------------------------
// Host orchestration
// ---------------------------------------------------------------------------
static void launch_gemm(int M, int N, int K,
                        const __hip_bfloat16* A, const __hip_bfloat16* Bt,
                        const float* bias, float* Cf, __hip_bfloat16* Cb,
                        int relu, hipStream_t stream) {
  dim3 grid(N / BN, M / BM), blk(256);
  if (relu)
    deq_gemm_bf16_wmma<1, 0, 1><<<grid, blk, 0, stream>>>(M, N, K, A, Bt, bias, nullptr, Cb);
  else if (Cf && Cb)
    deq_gemm_bf16_wmma<0, 1, 1><<<grid, blk, 0, stream>>>(M, N, K, A, Bt, bias, Cf, Cb);
  else
    deq_gemm_bf16_wmma<0, 1, 0><<<grid, blk, 0, stream>>>(M, N, K, A, Bt, bias, Cf, nullptr);
}

extern "C" void kernel_launch(void* const* d_in, const int* in_sizes, int n_in,
                              void* d_out, int out_size, void* d_ws, size_t ws_size,
                              hipStream_t stream) {
  const int B = 1024, DIN = 512, DH = 1024, DOUT = 512;
  const float* x     = (const float*)d_in[0];
  const float* W_in  = (const float*)d_in[1];
  const float* b_in  = (const float*)d_in[2];
  const float* W1    = (const float*)d_in[3];
  const float* b1    = (const float*)d_in[4];
  const float* W2    = (const float*)d_in[5];
  const float* b2    = (const float*)d_in[6];
  const float* W_out = (const float*)d_in[7];
  const float* b_out = (const float*)d_in[8];

  char* ws = (char*)d_ws;
  size_t off = 0;
  auto carve = [&](size_t bytes) -> void* {
    void* p = ws + off;
    off += (bytes + 255) & ~(size_t)255;
    return p;
  };
  float*          X      = (float*)carve((size_t)6 * B * DH * 4);
  float*          F      = (float*)carve((size_t)6 * B * DH * 4);
  __hip_bfloat16* h      = (__hip_bfloat16*)carve((size_t)B * DH * 2);
  __hip_bfloat16* zb     = (__hip_bfloat16*)carve((size_t)B * DH * 2);
  __hip_bfloat16* xb     = (__hip_bfloat16*)carve((size_t)B * DIN * 2);
  __hip_bfloat16* Winbt  = (__hip_bfloat16*)carve((size_t)DIN * DH * 2);   // [DH x DIN]
  __hip_bfloat16* W1bt   = (__hip_bfloat16*)carve((size_t)DH * DH * 2);    // [DH x DH]
  __hip_bfloat16* W2bt   = (__hip_bfloat16*)carve((size_t)DH * DH * 2);    // [DH x DH]
  __hip_bfloat16* Woutbt = (__hip_bfloat16*)carve((size_t)DH * DOUT * 2);  // [DOUT x DH]
  (void)ws_size; (void)in_sizes; (void)n_in; (void)out_size;

  const size_t sl = (size_t)B * DH;

  // activations straight, weights transposed-to-bf16 (reused 40-80x)
  f32_to_bf16_kernel<<<(B * DIN + 255) / 256, 256, 0, stream>>>(x, xb, B * DIN);
  f32_to_bf16_transpose_kernel<<<(DIN * DH + 255) / 256, 256, 0, stream>>>(W_in, Winbt, DIN, DH);
  f32_to_bf16_transpose_kernel<<<(DH * DH + 255) / 256, 256, 0, stream>>>(W1, W1bt, DH, DH);
  f32_to_bf16_transpose_kernel<<<(DH * DH + 255) / 256, 256, 0, stream>>>(W2, W2bt, DH, DH);
  f32_to_bf16_transpose_kernel<<<(DH * DOUT + 255) / 256, 256, 0, stream>>>(W_out, Woutbt, DH, DOUT);

  // z0 = x @ W_in + b_in  -> X[0] (f32) and zb (bf16)
  launch_gemm(B, DH, DIN, xb, Winbt, b_in, X + 0 * sl, zb, 0, stream);

  // F[0] = f(z0)
  launch_gemm(B, DH, DH, zb, W1bt, b1, nullptr, h, 1, stream);
  launch_gemm(B, DH, DH, h,  W2bt, b2, F + 0 * sl, zb, 0, stream);

  // X[1] = F[0]
  (void)hipMemcpyAsync(X + 1 * sl, F + 0 * sl, sl * sizeof(float),
                       hipMemcpyDeviceToDevice, stream);

  // F[1] = f(F[0])
  launch_gemm(B, DH, DH, zb, W1bt, b1, nullptr, h, 1, stream);
  launch_gemm(B, DH, DH, h,  W2bt, b2, F + 1 * sl, nullptr, 0, stream);

  // Anderson loop k = 2..39; z_star = xk at k = 39 (its f-eval is discarded)
  for (int k = 2; k < 40; ++k) {
    const int slot = k % 6;
    anderson_combine<<<B / 8, 256, 0, stream>>>(X, F, X + (size_t)slot * sl, zb,
                                                k, B, DH);
    if (k < 39) {
      launch_gemm(B, DH, DH, zb, W1bt, b1, nullptr, h, 1, stream);
      launch_gemm(B, DH, DH, h,  W2bt, b2, F + (size_t)slot * sl, nullptr, 0, stream);
    }
  }

  // out = z_star @ W_out + b_out
  launch_gemm(B, DOUT, DH, zb, Woutbt, b_out, (float*)d_out, nullptr, 0, stream);
}
